// GearNet_60705067762189
// MI455X (gfx1250) — compile-verified, matrix operand order checked by reading, not run on hardware
//
#include <hip/hip_runtime.h>
#include <hip/hip_bf16.h>

typedef __attribute__((ext_vector_type(16))) __bf16 v16bf;
typedef __attribute__((ext_vector_type(8)))  __bf16 v8bf;
typedef __attribute__((ext_vector_type(8)))  float  v8f;

#define R_   7
#define N_   50000
#define E_   800000
#define DIN_ 128
#define D_   256
#define G_   64
#define KE_  265
#define KEP_ 288   // 265 padded to multiple of 32 for bf16 WMMA K

// ---------------------------------------------------------------------------
// WMMA fragment helpers (CDNA5 16-bit A/B layout, ISA 7.12.2):
// lane L: row = L%16, k-base = (L/16)*8 ; elements {kb..kb+7, kb+16..kb+23}
// Works for global or LDS pointers (addrspace inferred after inlining).
// ---------------------------------------------------------------------------
__device__ __forceinline__ v16bf load_frag16(const __bf16* p) {
  v8bf lo = *(const v8bf*)(p);
  v8bf hi = *(const v8bf*)(p + 16);
  v16bf r;
#pragma unroll
  for (int i = 0; i < 8; ++i) { r[i] = lo[i]; r[i + 8] = hi[i]; }
  return r;
}

__device__ __forceinline__ v8f wmma_bf16(v16bf a, v16bf b, v8f c) {
  // D = A(16x32 bf16) * B(32x16 bf16) + C(16x16 f32)
  return __builtin_amdgcn_wmma_f32_16x16x32_bf16(false, a, false, b, (short)0, c,
                                                 false, false);
}

__device__ __forceinline__ v8f zero8() {
  v8f z;
#pragma unroll
  for (int i = 0; i < 8; ++i) z[i] = 0.0f;
  return z;
}

// ---------------------------------------------------------------------------
// Weight prep: f32 [K][Nn] row-major  ->  bf16 [Nn][Kpad] (zero padded K)
// ---------------------------------------------------------------------------
__global__ void k_transpose_cvt(const float* __restrict__ in, __bf16* __restrict__ out,
                                int K, int Nn, int Kpad) {
  int t = blockIdx.x * 256 + threadIdx.x;
  if (t >= Nn * Kpad) return;
  int n = t / Kpad, k = t % Kpad;
  out[t] = (k < K) ? (__bf16)in[(size_t)k * Nn + n] : (__bf16)0.0f;
}

// ---------------------------------------------------------------------------
// Edge features: [x[src](128) | x[dst](128) | onehot(7) | seqdist | dist | 0pad]
// bf16, fixed across all layers.  One 288-thread (9-wave) block per edge.
// ---------------------------------------------------------------------------
__global__ void k_edge_feat(const float* __restrict__ x, const float* __restrict__ pos,
                            const int* __restrict__ src, const int* __restrict__ dst,
                            const int* __restrict__ et, __bf16* __restrict__ ef) {
  size_t e = blockIdx.x;
  int c = threadIdx.x;
  int s = src[e], t = dst[e];
  float v;
  if (c < 128)       v = x[(size_t)s * DIN_ + c];
  else if (c < 256)  v = x[(size_t)t * DIN_ + (c - 128)];
  else if (c < 263)  v = (et[e] == c - 256) ? 1.0f : 0.0f;
  else if (c == 263) v = fabsf((float)(s - t));
  else if (c == 264) {
    float dx = pos[(size_t)s * 3 + 0] - pos[(size_t)t * 3 + 0] + 1e-6f;
    float dy = pos[(size_t)s * 3 + 1] - pos[(size_t)t * 3 + 1] + 1e-6f;
    float dz = pos[(size_t)s * 3 + 2] - pos[(size_t)t * 3 + 2] + 1e-6f;
    v = sqrtf(dx * dx + dy * dy + dz * dz);
  } else v = 0.0f;
  ef[e * KEP_ + c] = (__bf16)v;
}

// ---------------------------------------------------------------------------
// elementwise f32 -> bf16
// ---------------------------------------------------------------------------
__global__ void k_cvt_bf16(const float* __restrict__ in, __bf16* __restrict__ out,
                           size_t n) {
  size_t i = (size_t)blockIdx.x * 256 + threadIdx.x;
  if (i < n) out[i] = (__bf16)in[i];
}

// ---------------------------------------------------------------------------
// Edge GEMM:  C[E, DD] = ef[E, 288] @ WtT + bias     (Wt stored [DD][288] bf16)
// block = 8 waves covering 64 rows x DD cols; wave = 16 rows x DD/2 cols.
// Weight matrix is staged in LDS in K-chunks of 96 (3 chunks); B-fragments
// are ds_loads, A is streamed from global with the chunk unrolled so the
// compiler can clause A loads ahead of the 3*NT-WMMA group.
// ---------------------------------------------------------------------------
template <int DD>
__global__ void __launch_bounds__(256)
k_gemm_edge(const __bf16* __restrict__ ef, const __bf16* __restrict__ Wt,
            const float* __restrict__ bias, float* __restrict__ C) {
  constexpr int NT = DD / 32;              // 16-col tiles per wave (4 or 8)
  constexpr int CH = 96;                   // K-chunk held in LDS (288 = 3*96)
  constexpr int SEGS = CH / 8;             // uint4 segments per row (12)
  __shared__ __bf16 sW[DD * CH];           // 24KB (DD=128) / 48KB (DD=256)

  const int tid   = threadIdx.x;
  const int lane  = tid & 31;
  const int wave  = tid >> 5;
  const size_t mBase = (size_t)blockIdx.x * 64 + (size_t)(wave & 3) * 16;
  const int nBase = (wave >> 2) * (DD / 2);
  const int row = lane & 15;
  const int kb  = (lane >> 4) * 8;

  v8f acc[NT];
#pragma unroll
  for (int i = 0; i < NT; ++i) acc[i] = zero8();

  const __bf16* aRow = ef + (mBase + row) * (size_t)KEP_ + kb;

  for (int kc = 0; kc < KEP_; kc += CH) {
    __syncthreads();
    // cooperative fill: DD rows x SEGS uint4 per row
    for (int i = tid; i < DD * SEGS; i += 256) {
      int r = i / SEGS, s = i % SEGS;
      ((uint4*)sW)[r * SEGS + s] =
          ((const uint4*)(Wt + (size_t)r * KEP_ + kc))[s];
    }
    __syncthreads();
#pragma unroll
    for (int k2 = 0; k2 < CH; k2 += 32) {
      v16bf a = load_frag16(aRow + kc + k2);
#pragma unroll
      for (int nt = 0; nt < NT; ++nt) {
        v16bf b = load_frag16(&sW[(nBase + nt * 16 + row) * CH + k2 + kb]);
        acc[nt] = wmma_bf16(a, b, acc[nt]);
      }
    }
  }
  // D layout: VGPR i -> M = i + 8*(lane>=16), N = lane%16
  const int mOff = (lane >> 4) * 8;
#pragma unroll
  for (int nt = 0; nt < NT; ++nt) {
    int col = nBase + nt * 16 + (lane & 15);
    float bb = bias[col];
#pragma unroll
    for (int i = 0; i < 8; ++i) {
      size_t m = mBase + mOff + i;
      C[m * DD + col] = acc[nt][i] + bb;
    }
  }
}

// ---------------------------------------------------------------------------
// Relation-typed scatter:  upd[(dst*R+et)*d + c] += Cedge[e,c] + h[src,c]
// fp32 atomics == reference segment_sum semantics.
// ---------------------------------------------------------------------------
__global__ void k_scatter(const float* __restrict__ Cedge, const float* __restrict__ h,
                          const int* __restrict__ src, const int* __restrict__ dst,
                          const int* __restrict__ et, float* __restrict__ upd, int d) {
  size_t e = blockIdx.x;
  int c = threadIdx.x;                       // blockDim == d
  int s = src[e], t = dst[e], r = et[e];
  float v = Cedge[e * (size_t)d + c] + h[(size_t)s * d + c];
  atomicAdd(upd + ((size_t)t * R_ + r) * d + c, v);
}

// ---------------------------------------------------------------------------
// Node GEMM:  h_new = relu(upd[N,K1]@lWt + lb + h[N,K2]@sWt + sb) (+ h residual)
// block = 16-row tile x 256 cols; 8 waves x 2 col-tiles each.
// The A tile (shared by all 8 waves) is staged in LDS in K-chunks of 128
// (16x128 bf16 = 4KB = exactly one uint4 per thread); B rows are per-wave
// distinct and streamed from global, 4 K-steps unrolled for clausing.
// ---------------------------------------------------------------------------
template <int K1, int K2, bool RES>
__global__ void __launch_bounds__(256)
k_gemm_node(const __bf16* __restrict__ updb, const __bf16* __restrict__ lWt,
            const float* __restrict__ lb, const __bf16* __restrict__ hb,
            const __bf16* __restrict__ sWt, const float* __restrict__ sb,
            const float* __restrict__ hres, float* __restrict__ hout,
            __bf16* __restrict__ houtb) {
  constexpr int CH = 128;                  // K-chunk staged in LDS
  __shared__ __bf16 sA[16 * CH];           // 4KB

  const int tid  = threadIdx.x;
  const int lane = tid & 31;
  const int wave = tid >> 5;
  const size_t mBase = (size_t)blockIdx.x * 16;
  const int n0  = wave * 32;
  const int row = lane & 15;
  const int kb  = (lane >> 4) * 8;
  const int fr = tid >> 4;                 // fill row (0..15)
  const int fs = tid & 15;                 // fill uint4 segment (0..15)

  v8f acc[2] = {zero8(), zero8()};

  // ---- pass 1: upd @ lWt ----
  for (int kc = 0; kc < K1; kc += CH) {
    __syncthreads();
    ((uint4*)sA)[tid] =
        ((const uint4*)(updb + (mBase + fr) * (size_t)K1 + kc))[fs];
    __syncthreads();
#pragma unroll
    for (int k2 = 0; k2 < CH; k2 += 32) {
      v16bf a = load_frag16(&sA[row * CH + k2 + kb]);
#pragma unroll
      for (int nt = 0; nt < 2; ++nt) {
        v16bf b = load_frag16(lWt + (size_t)(n0 + nt * 16 + row) * K1 + kc + k2 + kb);
        acc[nt] = wmma_bf16(a, b, acc[nt]);
      }
    }
  }
  // ---- pass 2: h @ sWt ----
  for (int kc = 0; kc < K2; kc += CH) {
    __syncthreads();
    ((uint4*)sA)[tid] =
        ((const uint4*)(hb + (mBase + fr) * (size_t)K2 + kc))[fs];
    __syncthreads();
#pragma unroll
    for (int k2 = 0; k2 < CH; k2 += 32) {
      v16bf a = load_frag16(&sA[row * CH + k2 + kb]);
#pragma unroll
      for (int nt = 0; nt < 2; ++nt) {
        v16bf b = load_frag16(sWt + (size_t)(n0 + nt * 16 + row) * K2 + kc + k2 + kb);
        acc[nt] = wmma_bf16(a, b, acc[nt]);
      }
    }
  }
  // ---- epilogue: bias + relu (+ residual), emit f32 and bf16 copies ----
  const int mOff = (lane >> 4) * 8;
#pragma unroll
  for (int nt = 0; nt < 2; ++nt) {
    int col = n0 + nt * 16 + (lane & 15);
    float bb = lb[col] + sb[col];
#pragma unroll
    for (int i = 0; i < 8; ++i) {
      size_t m = mBase + mOff + i;
      float v = acc[nt][i] + bb;
      v = v > 0.0f ? v : 0.0f;          // relu
      if (RES) v += hres[m * D_ + col]; // short-cut residual
      hout[m * D_ + col]  = v;
      houtb[m * D_ + col] = (__bf16)v;
    }
  }
}

// ---------------------------------------------------------------------------
// Graph readout: graph_emb[batch[n], c] += h[n, c]
// ---------------------------------------------------------------------------
__global__ void k_readout(const float* __restrict__ h, const int* __restrict__ batch,
                          float* __restrict__ g) {
  size_t n = blockIdx.x;
  int c = threadIdx.x;  // 256
  atomicAdd(g + (size_t)batch[n] * D_ + c, h[n * (size_t)D_ + c]);
}

// ---------------------------------------------------------------------------
extern "C" void kernel_launch(void* const* d_in, const int* in_sizes, int n_in,
                              void* d_out, int out_size, void* d_ws, size_t ws_size,
                              hipStream_t stream) {
  (void)in_sizes; (void)n_in; (void)out_size; (void)ws_size;
  const float* x    = (const float*)d_in[0];
  const float* pos  = (const float*)d_in[1];
  const int*   ei   = (const int*)d_in[2];
  const int*   et   = (const int*)d_in[3];
  const int*   batch= (const int*)d_in[4];
  const float* l0eW = (const float*)d_in[5];
  const float* l0eb = (const float*)d_in[6];
  const float* l0lW = (const float*)d_in[7];
  const float* l0lb = (const float*)d_in[8];
  const float* l0sW = (const float*)d_in[9];
  const float* l0sb = (const float*)d_in[10];
  const float* eW   = (const float*)d_in[11];
  const float* ebt  = (const float*)d_in[12];
  const float* lW   = (const float*)d_in[13];
  const float* lbt  = (const float*)d_in[14];
  const float* sW   = (const float*)d_in[15];
  const float* sbt  = (const float*)d_in[16];
  const int* src = ei;
  const int* dst = ei + E_;

  char* ws = (char*)d_ws;
  size_t off = 0;
  auto alloc = [&](size_t bytes) -> void* {
    off = (off + 255) & ~(size_t)255;
    void* p = ws + off; off += bytes; return p;
  };
  __bf16* ef    = (__bf16*)alloc((size_t)E_ * KEP_ * 2);
  float*  Cedge = (float*) alloc((size_t)E_ * D_ * 4);
  float*  upd   = (float*) alloc((size_t)N_ * R_ * D_ * 4);
  __bf16* updb  = (__bf16*)alloc((size_t)N_ * R_ * D_ * 2);
  float*  hA    = (float*) alloc((size_t)N_ * D_ * 4);
  float*  hB    = (float*) alloc((size_t)N_ * D_ * 4);
  __bf16* hbf0  = (__bf16*)alloc((size_t)N_ * D_ * 2);
  __bf16* hbf1  = (__bf16*)alloc((size_t)N_ * D_ * 2);
  __bf16* eWt0  = (__bf16*)alloc((size_t)DIN_ * KEP_ * 2);
  __bf16* eWt   = (__bf16*)alloc((size_t)5 * D_ * KEP_ * 2);
  __bf16* lWt0  = (__bf16*)alloc((size_t)D_ * (R_ * DIN_) * 2);
  __bf16* lWt   = (__bf16*)alloc((size_t)5 * D_ * (R_ * D_) * 2);
  __bf16* sWt0  = (__bf16*)alloc((size_t)D_ * DIN_ * 2);
  __bf16* sWt   = (__bf16*)alloc((size_t)5 * D_ * D_ * 2);

  // ---- weight prep: transpose + bf16 ----
  k_transpose_cvt<<<(DIN_ * KEP_ + 255) / 256, 256, 0, stream>>>(l0eW, eWt0, KE_, DIN_, KEP_);
  for (int i = 0; i < 5; ++i)
    k_transpose_cvt<<<(D_ * KEP_ + 255) / 256, 256, 0, stream>>>(
        eW + (size_t)i * KE_ * D_, eWt + (size_t)i * D_ * KEP_, KE_, D_, KEP_);
  k_transpose_cvt<<<(D_ * R_ * DIN_ + 255) / 256, 256, 0, stream>>>(l0lW, lWt0, R_ * DIN_, D_, R_ * DIN_);
  for (int i = 0; i < 5; ++i)
    k_transpose_cvt<<<(D_ * R_ * D_ + 255) / 256, 256, 0, stream>>>(
        lW + (size_t)i * R_ * D_ * D_, lWt + (size_t)i * D_ * R_ * D_, R_ * D_, D_, R_ * D_);
  k_transpose_cvt<<<(D_ * DIN_ + 255) / 256, 256, 0, stream>>>(l0sW, sWt0, DIN_, D_, DIN_);
  for (int i = 0; i < 5; ++i)
    k_transpose_cvt<<<(D_ * D_ + 255) / 256, 256, 0, stream>>>(
        sW + (size_t)i * D_ * D_, sWt + (size_t)i * D_ * D_, D_, D_, D_);

  // ---- edge features once; x -> bf16 for layer-0 self GEMM ----
  k_edge_feat<<<E_, KEP_, 0, stream>>>(x, pos, src, dst, et, ef);
  k_cvt_bf16<<<(N_ * DIN_ + 255) / 256, 256, 0, stream>>>(x, hbf0, (size_t)N_ * DIN_);

  // ---- layer 0 (d_in = 128 -> 256, no residual) ----
  k_gemm_edge<DIN_><<<E_ / 64, 256, 0, stream>>>(ef, eWt0, l0eb, Cedge);
  hipMemsetAsync(upd, 0, (size_t)N_ * R_ * DIN_ * 4, stream);
  k_scatter<<<E_, DIN_, 0, stream>>>(Cedge, x, src, dst, et, upd, DIN_);
  k_cvt_bf16<<<(N_ * R_ * DIN_ + 255) / 256, 256, 0, stream>>>(upd, updb, (size_t)N_ * R_ * DIN_);
  k_gemm_node<R_ * DIN_, DIN_, false><<<N_ / 16, 256, 0, stream>>>(
      updb, lWt0, l0lb, hbf0, sWt0, l0sb, nullptr, hA, hbf1);

  // ---- layers 1..5 (256 -> 256, residual); layer 5 writes h into d_out ----
  float*  hcur  = hA;
  float*  hnext = hB;
  __bf16* hbcur = hbf1;
  __bf16* hbnext = hbf0;
  for (int i = 0; i < 5; ++i) {
    k_gemm_edge<D_><<<E_ / 64, 256, 0, stream>>>(
        ef, eWt + (size_t)i * D_ * KEP_, ebt + (size_t)i * D_, Cedge);
    hipMemsetAsync(upd, 0, (size_t)N_ * R_ * D_ * 4, stream);
    k_scatter<<<E_, D_, 0, stream>>>(Cedge, hcur, src, dst, et, upd, D_);
    k_cvt_bf16<<<(N_ * R_ * D_ + 255) / 256, 256, 0, stream>>>(upd, updb, (size_t)N_ * R_ * D_);
    float* outp = (i == 4) ? (float*)d_out : hnext;
    k_gemm_node<R_ * D_, D_, true><<<N_ / 16, 256, 0, stream>>>(
        updb, lWt + (size_t)i * D_ * R_ * D_, lbt + (size_t)i * D_, hbcur,
        sWt + (size_t)i * D_ * D_, sbt + (size_t)i * D_, hcur, outp, hbnext);
    float* tf = hcur; hcur = outp; hnext = tf;
    __bf16* tb = hbcur; hbcur = hbnext; hbnext = tb;
  }

  // ---- graph readout into tail of d_out ----
  float* gout = (float*)d_out + (size_t)N_ * D_;
  hipMemsetAsync(gout, 0, (size_t)G_ * D_ * 4, stream);
  k_readout<<<N_, D_, 0, stream>>>((const float*)d_out, batch, gout);
}
